// ModulatedConv2d_1434519077119
// MI455X (gfx1250) — compile-verified, matrix operand order checked by reading, not run on hardware
//
#include <hip/hip_runtime.h>
#include <math.h>

// ---------------- problem constants ----------------
#define BATCH 16
#define CIN   512
#define COUT  512
#define SDIM  512
#define HH    64
#define WW    64
#define KTOT  (9 * CIN)             // 4608, K-dim of implicit GEMM, order (kh,kw,ci)

#define MOD_SCALE_F  0.04419417382f     // 1/sqrt(512)
#define CONV_SCALE_F 0.014731391f       // 1/sqrt(512*9)
#define EPS_F        1e-8f

typedef __attribute__((ext_vector_type(16))) _Float16 v16h;
typedef __attribute__((ext_vector_type(8)))  _Float16 v8h;
typedef __attribute__((ext_vector_type(8)))  float    v8f;

// ---------------- small prep kernels ----------------

// s[b,ci] = sum_d style[b,d] * mod_weight[ci,d] * MOD_SCALE + mod_bias[ci]
__global__ void k_style_mod(const float* __restrict__ style,
                            const float* __restrict__ mw,
                            const float* __restrict__ mb,
                            float* __restrict__ s) {
    int idx = blockIdx.x * blockDim.x + threadIdx.x;
    if (idx >= BATCH * CIN) return;
    int b = idx / CIN, ci = idx % CIN;
    const float* st = style + (size_t)b * SDIM;
    const float* w  = mw + (size_t)ci * SDIM;
    float acc = 0.f;
    for (int d = 0; d < SDIM; ++d) acc = fmaf(st[d], w[d], acc);
    s[idx] = acc * MOD_SCALE_F + mb[ci];
}

// wsum[co,ci] = sum_k conv_weight[co,ci,k]^2
__global__ void k_wsum(const float* __restrict__ cw, float* __restrict__ wsum) {
    int idx = blockIdx.x * blockDim.x + threadIdx.x;
    if (idx >= COUT * CIN) return;
    const float* p = cw + (size_t)idx * 9;
    float a = 0.f;
#pragma unroll
    for (int k = 0; k < 9; ++k) a = fmaf(p[k], p[k], a);
    wsum[idx] = a;
}

// demod[b,co] = rsqrt(CONV_SCALE^2 * sum_ci s^2 * wsum + eps)
__global__ void k_demod(const float* __restrict__ s,
                        const float* __restrict__ wsum,
                        float* __restrict__ demod) {
    int idx = blockIdx.x * blockDim.x + threadIdx.x;
    if (idx >= BATCH * COUT) return;
    int b = idx / COUT, co = idx % COUT;
    const float* sb = s + (size_t)b * CIN;
    const float* wr = wsum + (size_t)co * CIN;
    float acc = 0.f;
    for (int ci = 0; ci < CIN; ++ci) acc = fmaf(sb[ci] * sb[ci], wr[ci], acc);
    demod[idx] = rsqrtf(CONV_SCALE_F * CONV_SCALE_F * acc + EPS_F);
}

// Aw[co][kid][ci] = f16(conv_weight[co][ci][kid])   (K reorder: kid outer, ci inner)
__global__ void k_pack_w(const float* __restrict__ cw, _Float16* __restrict__ Aw) {
    int idx = blockIdx.x * blockDim.x + threadIdx.x;
    if (idx >= COUT * KTOT) return;
    int co  = idx / KTOT;
    int r   = idx % KTOT;
    int kid = r / CIN;
    int ci  = r % CIN;
    Aw[idx] = (_Float16)cw[((size_t)co * CIN + ci) * 9 + kid];
}

// xm[b,ci,h,w] = f16(s[b,ci] * input[b,ci,h,w])
__global__ void k_xmod(const float* __restrict__ inp,
                       const float* __restrict__ s,
                       _Float16* __restrict__ xm) {
    int idx = blockIdx.x * blockDim.x + threadIdx.x;
    if (idx >= BATCH * CIN * HH * WW) return;
    int bci = idx / (HH * WW);          // b*CIN + ci
    xm[idx] = (_Float16)(inp[idx] * s[bci]);
}

// ---------------- main implicit-GEMM WMMA conv ----------------
// grid: 16(b) * 4(co-tile 128) * 32(row-pairs) = 2048 blocks, 256 threads (8 wave32)
// block tile: M=128 (co), N=128 (two image rows), K = 9*512 in 32-chunks
// waves: 2 along M (64 rows) x 4 along N (32 cols) -> 8 wmma per wave per K-step
// A tiles: triple-buffered via global_load_async_to_lds_b128, partial asynccnt waits
__global__ __launch_bounds__(256) void k_conv_wmma(
    const _Float16* __restrict__ Aw,     // [COUT][9][CIN] f16
    const _Float16* __restrict__ xm,     // [B][CIN][H][W] f16
    const float*    __restrict__ demod,  // [B][COUT]
    float*          __restrict__ out) {  // [B][COUT][H][W]
    __shared__ _Float16 As[3][128 * 32]; // triple-buffered A tile, async-filled (24 KB)
    __shared__ _Float16 Bsr[4 * 64 * 32];// [halo row r][w][kk] per ci-chunk (16 KB)

    const int tid   = threadIdx.x;
    const int lane  = tid & 31;
    const int wave  = tid >> 5;
    const int waveN = wave & 3;          // 4 waves along N (32 cols each)
    const int waveM = wave >> 2;         // 2 waves along M (64 rows each)
    const int lhalf = lane >> 4;         // 0: lanes 0-15, 1: lanes 16-31
    const int lmod  = lane & 15;

    const int bid = blockIdx.x;
    const int h0  = (bid & 31) * 2;      // first of two output rows
    const int ct  = (bid >> 5) & 3;      // co tile
    const int b   = bid >> 7;            // batch
    const int co0 = ct * 128;

    const int nrowW  = waveN >> 1;       // which of the 2 output rows this wave does
    const int wbase0 = (waveN & 1) * 32; // col base within that row

    const int arow0  = tid >> 2;         // A staging: 2 chunks/thread
    const int achunk = tid & 3;

    v8f acc[4][2] = {};

    // ---- async-stage the A tile (128x32 f16) into LDS buffer pb ----
    // each wave issues exactly 2 async instructions per tile (ordered completion)
    auto issueA = [&](int cc, int kid, int pb) {
        const _Float16* g0 = Aw + (size_t)co0 * KTOT + kid * CIN + cc * 32;
#pragma unroll
        for (int i = 0; i < 2; ++i) {
            int row = arow0 + i * 64;
            const _Float16* g = g0 + (size_t)row * KTOT + achunk * 8;
            unsigned ldsoff = (unsigned)(uintptr_t)(&As[pb][row * 32 + achunk * 8]);
            asm volatile("global_load_async_to_lds_b128 %0, %1, off"
                         :: "v"(ldsoff), "v"(g) : "memory");
        }
    };

    // ---- stage 4 halo rows of modulated input, transposed to [r][w][kk] ----
    auto stageB = [&](int cc) {
        const _Float16* xb = xm + ((size_t)b * CIN + cc * 32) * (HH * WW);
#pragma unroll
        for (int e = 0; e < 16; ++e) {
            int j   = tid + e * 256;     // 0..4095 u32 units
            int r   = j >> 10;           // halo row 0..3
            int rem = j & 1023;
            int kk  = rem >> 5;          // 0..31 (ci within chunk)
            int wp  = rem & 31;          // w-pair, coalesced along tid
            int hr  = h0 - 1 + r;
            unsigned u = 0u;
            if (hr >= 0 && hr < HH)
                u = *(const unsigned*)(xb + (size_t)kk * (HH * WW) + hr * WW + wp * 2);
            union { unsigned u32; _Float16 h[2]; } cv; cv.u32 = u;
            Bsr[((r * 64 + wp * 2)     * 32) + kk] = cv.h[0];
            Bsr[((r * 64 + wp * 2 + 1) * 32) + kk] = cv.h[1];
        }
    };

    // ---- one K-step: build fragments, 8 wmma ----
    auto compute = [&](int kid, int pb) {
        const int kh = kid / 3, kw = kid % 3;
        // A fragments (CDNA5 16-bit A layout: lane<16 K{0-7,16-23}, lane>=16 K{8-15,24-31})
        v16h af[4];
#pragma unroll
        for (int mi = 0; mi < 4; ++mi) {
            int row = waveM * 64 + mi * 16 + lmod;
            v8h lo = *(const v8h*)(&As[pb][row * 32 + lhalf * 8]);
            v8h hi = *(const v8h*)(&As[pb][row * 32 + 16 + lhalf * 8]);
            af[mi] = __builtin_shufflevector(lo, hi, 0,1,2,3,4,5,6,7,8,9,10,11,12,13,14,15);
        }
        // B fragments: kw shift applied at load time, zero-masked at edges
        v16h bf[2];
        const int r = nrowW + kh;        // uniform per wave
#pragma unroll
        for (int ni = 0; ni < 2; ++ni) {
            int w  = wbase0 + ni * 16 + lmod;
            int m  = w + kw - 1;
            int mc = m < 0 ? 0 : (m > 63 ? 63 : m);
            v8h lo = *(const v8h*)(&Bsr[(r * 64 + mc) * 32 + lhalf * 8]);
            v8h hi = *(const v8h*)(&Bsr[(r * 64 + mc) * 32 + 16 + lhalf * 8]);
            if (m != mc) { lo = 0; hi = 0; }   // per-lane select, EXEC stays full
            bf[ni] = __builtin_shufflevector(lo, hi, 0,1,2,3,4,5,6,7,8,9,10,11,12,13,14,15);
        }
#pragma unroll
        for (int mi = 0; mi < 4; ++mi)
#pragma unroll
            for (int ni = 0; ni < 2; ++ni)
                acc[mi][ni] = __builtin_amdgcn_wmma_f32_16x16x32_f16(
                    false, af[mi], false, bf[ni], (short)0, acc[mi][ni], false, false);
    };

    // ---- pipeline: 2 A tiles in flight, B halo staged per ci-chunk ----
    issueA(0, 0, 0);
    issueA(0, 1, 1);
    stageB(0);

    for (int cc = 0; cc < 16; ++cc) {
#pragma unroll
        for (int kid = 0; kid < 9; ++kid) {
            // tile t = cc*9+kid must be resident; tile t+1 may still be in flight
            if (cc == 15 && kid == 8)
                asm volatile("s_wait_asynccnt 0x0" ::: "memory");
            else
                asm volatile("s_wait_asynccnt 0x2" ::: "memory");
            __syncthreads();   // also retires all readers of buffer (t+2)%3 = (t-1)%3

            // prefetch tile t+2 (indices constant-folded per unrolled kid)
            if (!(cc == 15 && kid >= 7)) {
                int cc2  = (kid + 2 < 9) ? cc : cc + 1;
                int kid2 = (kid + 2 < 9) ? kid + 2 : kid - 7;
                issueA(cc2, kid2, (kid + 2) % 3);
            }
            // new ci-chunk: restage the B halo (readers of old Bsr retired above)
            if (kid == 0 && cc > 0) {
                stageB(cc);
                __syncthreads();
            }
            compute(kid, kid % 3);
        }
    }

    // ---- epilogue: scale by demod*CONV_SCALE, store f32 ----
    // C/D layout: VGPR r holds M=r (lanes 0-15) / M=r+8 (lanes 16-31), N=lane%16
#pragma unroll
    for (int mi = 0; mi < 4; ++mi) {
#pragma unroll
        for (int rr = 0; rr < 8; ++rr) {
            int   co    = co0 + waveM * 64 + mi * 16 + rr + lhalf * 8;
            float scale = demod[(size_t)b * COUT + co] * CONV_SCALE_F;
#pragma unroll
            for (int ni = 0; ni < 2; ++ni) {
                int w = wbase0 + ni * 16 + lmod;
                out[((size_t)(b * COUT + co) * HH + (h0 + nrowW)) * WW + w] =
                    acc[mi][ni][rr] * scale;
            }
        }
    }
}

// ---------------- launcher ----------------
extern "C" void kernel_launch(void* const* d_in, const int* in_sizes, int n_in,
                              void* d_out, int out_size, void* d_ws, size_t ws_size,
                              hipStream_t stream) {
    const float* inp   = (const float*)d_in[0];   // [16,512,64,64]
    const float* style = (const float*)d_in[1];   // [16,512]
    const float* cw    = (const float*)d_in[2];   // [1,512,512,3,3]
    const float* mw    = (const float*)d_in[3];   // [512,512]
    const float* mb    = (const float*)d_in[4];   // [512]
    float* out = (float*)d_out;

    // workspace carve-up (all 16B-aligned by construction)
    float* s     = (float*)d_ws;                       // 16*512
    float* demod = s + BATCH * CIN;                    // 16*512
    float* wsum  = demod + BATCH * COUT;               // 512*512
    _Float16* Aw = (_Float16*)(wsum + COUT * CIN);     // 512*4608 halves (~4.7 MB)
    _Float16* xm = Aw + (size_t)COUT * KTOT;           // 16*512*4096 halves (~67 MB)

    k_style_mod<<<(BATCH * CIN + 255) / 256, 256, 0, stream>>>(style, mw, mb, s);
    k_wsum<<<(COUT * CIN + 255) / 256, 256, 0, stream>>>(cw, wsum);
    k_demod<<<(BATCH * COUT + 255) / 256, 256, 0, stream>>>(s, wsum, demod);
    k_pack_w<<<(COUT * KTOT + 255) / 256, 256, 0, stream>>>(cw, Aw);
    k_xmod<<<(BATCH * CIN * HH * WW + 255) / 256, 256, 0, stream>>>(inp, s, xm);
    k_conv_wmma<<<BATCH * 4 * 32, 256, 0, stream>>>(Aw, xm, demod, out);
}
